// MultiScaleSequenceScanner_6648609374656
// MI455X (gfx1250) — compile-verified, matrix operand order checked by reading, not run on hardware
//
#include <hip/hip_runtime.h>
#include <hip/hip_bf16.h>

typedef __attribute__((ext_vector_type(16))) _Float16 v16h;
typedef __attribute__((ext_vector_type(8)))  _Float16 v8h;
typedef __attribute__((ext_vector_type(8)))  float    v8f;

#define IMGW 224
#define PADW 232
#define DCH  512
#define KTOT 768   // 3*16*16

// ---------------------------------------------------------------------------
// Kernel 1: 3x3 deform conv (3->3 ch, pad 1) -> f16 "guided" in two layouts:
//   g  : (64,3,224,224)  for stride-16 embeds (patch x-origin = 16*pw, 8-aligned)
//   gp : (64,3,232,232)  zero-padded by 4   (patch x-origin = 8*pw, 8-aligned)
// ---------------------------------------------------------------------------
__global__ __launch_bounds__(256) void deform_conv_kernel(
    const float* __restrict__ img, const float* __restrict__ wd,
    _Float16* __restrict__ g, _Float16* __restrict__ gp)
{
    __shared__ float ws[81];
    if (threadIdx.x < 81) ws[threadIdx.x] = wd[threadIdx.x];
    __syncthreads();

    long idx = (long)blockIdx.x * blockDim.x + threadIdx.x;   // over padded domain
    int x = (int)(idx % PADW); long t = idx / PADW;
    int y = (int)(t % PADW);   t /= PADW;
    int c = (int)(t % 3);      int b = (int)(t / 3);

    int iy = y - 4, ix = x - 4;
    float v = 0.0f;
    if (iy >= 0 && iy < IMGW && ix >= 0 && ix < IMGW) {
        const float* ip = img + (long)b * 3 * IMGW * IMGW;
        #pragma unroll
        for (int ci = 0; ci < 3; ++ci)
            #pragma unroll
            for (int dy = 0; dy < 3; ++dy) {
                int yy = iy + dy - 1;
                if (yy < 0 || yy >= IMGW) continue;
                #pragma unroll
                for (int dx = 0; dx < 3; ++dx) {
                    int xx = ix + dx - 1;
                    if (xx < 0 || xx >= IMGW) continue;
                    v += ip[((long)ci * IMGW + yy) * IMGW + xx] * ws[((c * 3 + ci) * 3 + dy) * 3 + dx];
                }
            }
        g[(((long)b * 3 + c) * IMGW + iy) * IMGW + ix] = (_Float16)v;
    }
    gp[(((long)b * 3 + c) * PADW + y) * PADW + x] = (_Float16)v;
}

// ---------------------------------------------------------------------------
// Kernel 2: weights f32 -> f16 (layout (512,768) already matches im2col K order)
// ---------------------------------------------------------------------------
__global__ __launch_bounds__(256) void wcvt_kernel(
    const float* __restrict__ w0, const float* __restrict__ w1, const float* __restrict__ w2,
    _Float16* __restrict__ o0, _Float16* __restrict__ o1, _Float16* __restrict__ o2)
{
    int i = blockIdx.x * 256 + threadIdx.x;
    if (i < DCH * KTOT) {
        o0[i] = (_Float16)w0[i];
        o1[i] = (_Float16)w1[i];
        o2[i] = (_Float16)w2[i];
    }
}

// ---------------------------------------------------------------------------
// Kernel 3: WMMA GEMM  (M x 768) x (768 x 512).
// Per-wave 32x32 tile (4 accumulators), software-pipelined: iteration k+1's
// 8 b128 loads are issued before iteration k's 4 WMMAs so loads overlap math.
// Waves within a block share the B (weight) tile. All tiles full => EXEC all-1s.
// Epilogue: bias add + snake (mode 0) / reverse-snake (mode 1) scatter, or
// plain (mode 2) write of t2 f32 to workspace.
// ---------------------------------------------------------------------------
__global__ __launch_bounds__(128) void patch_gemm_kernel(
    const _Float16* __restrict__ A, const _Float16* __restrict__ Wm,
    const float* __restrict__ bias, float* __restrict__ out,
    int Himg, int stride, int gh, int gw, int mode)
{
    const int lane = threadIdx.x & 31;
    const int wave = threadIdx.x >> 5;
    const int NT = DCH / 32;                 // 16 n-tiles of 32 cols
    int nt  = blockIdx.x % NT;
    int mtg = blockIdx.x / NT;
    int mt  = mtg * 4 + wave;                // 32-row tile index (waves share nt)
    int m0 = mt * 32, n0 = nt * 32;

    int nl = lane & 15;                      // row (A) / column (B,C) within 16
    int kh = lane >> 4;                      // K-phase (0: K%16<8, 1: K%16>=8)
    int gsz = gh * gw;
    long planeHW = (long)Himg * Himg;

    // per-lane patch base pointers for the two 16-row A sub-tiles
    const _Float16* Abase[2];
    #pragma unroll
    for (int mi = 0; mi < 2; ++mi) {
        int m  = m0 + 16 * mi + nl;
        int b  = m / gsz;
        int p  = m - b * gsz;
        int ph = p / gw;
        int pw = p - ph * gw;
        Abase[mi] = A + (long)b * 3 * planeHW + (long)(ph * stride) * Himg + pw * stride;
    }
    const _Float16* Wb0 = Wm + (long)(n0 + nl) * KTOT + 8 * kh;
    const _Float16* Wb1 = Wb0 + 16 * KTOT;

    auto loadA = [&](const _Float16* base, int kb) -> v16h {
        int K0 = kb + 8 * kh, K1 = K0 + 16;
        v8h lo = *(const v8h*)(base + (K0 >> 8) * planeHW + ((K0 >> 4) & 15) * Himg + (K0 & 15));
        v8h hi = *(const v8h*)(base + (K1 >> 8) * planeHW + ((K1 >> 4) & 15) * Himg + (K1 & 15));
        return __builtin_shufflevector(lo, hi, 0,1,2,3,4,5,6,7,8,9,10,11,12,13,14,15);
    };
    auto loadB = [&](const _Float16* base, int kb) -> v16h {
        v8h lo = *(const v8h*)(base + kb);
        v8h hi = *(const v8h*)(base + kb + 16);
        return __builtin_shufflevector(lo, hi, 0,1,2,3,4,5,6,7,8,9,10,11,12,13,14,15);
    };

    v8f a00 = {}, a01 = {}, a10 = {}, a11 = {};

    // prologue loads
    v16h A0 = loadA(Abase[0], 0);
    v16h A1 = loadA(Abase[1], 0);
    v16h B0 = loadB(Wb0, 0);
    v16h B1 = loadB(Wb1, 0);

    for (int kb = 0; kb < KTOT - 32; kb += 32) {
        int kn = kb + 32;
        // issue next iteration's loads first -> overlap with this iteration's WMMAs
        v16h nA0 = loadA(Abase[0], kn);
        v16h nA1 = loadA(Abase[1], kn);
        v16h nB0 = loadB(Wb0, kn);
        v16h nB1 = loadB(Wb1, kn);
        __builtin_prefetch(Wb0 + kn + 32, 0, 1);           // global_prefetch_b8

        a00 = __builtin_amdgcn_wmma_f32_16x16x32_f16(false, A0, false, B0, (short)0, a00, false, false);
        a01 = __builtin_amdgcn_wmma_f32_16x16x32_f16(false, A0, false, B1, (short)0, a01, false, false);
        a10 = __builtin_amdgcn_wmma_f32_16x16x32_f16(false, A1, false, B0, (short)0, a10, false, false);
        a11 = __builtin_amdgcn_wmma_f32_16x16x32_f16(false, A1, false, B1, (short)0, a11, false, false);

        A0 = nA0; A1 = nA1; B0 = nB0; B1 = nB1;
    }
    // epilogue K-step
    a00 = __builtin_amdgcn_wmma_f32_16x16x32_f16(false, A0, false, B0, (short)0, a00, false, false);
    a01 = __builtin_amdgcn_wmma_f32_16x16x32_f16(false, A0, false, B1, (short)0, a01, false, false);
    a10 = __builtin_amdgcn_wmma_f32_16x16x32_f16(false, A1, false, B0, (short)0, a10, false, false);
    a11 = __builtin_amdgcn_wmma_f32_16x16x32_f16(false, A1, false, B1, (short)0, a11, false, false);

    float bv0 = bias[n0 + nl];
    float bv1 = bias[n0 + 16 + nl];

    // C/D layout: VGPR v, lanes 0-15 -> M=v, lanes 16-31 -> M=v+8; N = lane&15.
    #pragma unroll
    for (int mi = 0; mi < 2; ++mi) {
        #pragma unroll
        for (int v = 0; v < 8; ++v) {
            int mr = m0 + 16 * mi + v + 8 * kh;
            int bb = mr / gsz;
            int pp = mr - bb * gsz;
            int r  = pp / gw;
            int cc = pp - r * gw;
            long row;
            if (mode == 0) {                       // snake scatter
                int c2 = (r & 1) ? (gw - 1 - cc) : cc;
                row = (long)bb * gsz + r * gw + c2;
            } else if (mode == 1) {                // reverse-snake scatter
                int k2 = gh - 1 - r;
                int j2 = (k2 & 1) ? cc : (gw - 1 - cc);
                row = (long)bb * gsz + k2 * gw + j2;
            } else {                               // plain -> t2 (B,28,28,512)
                row = mr;
            }
            float* dst = out + row * DCH + n0 + nl;
            if (mi == 0) {
                dst[0]  = a00[v] + bv0;
                dst[16] = a01[v] + bv1;
            } else {
                dst[0]  = a10[v] + bv0;
                dst[16] = a11[v] + bv1;
            }
        }
    }
}

// ---------------------------------------------------------------------------
// Kernel 4: bilinear 28x28 -> 42x42 on (64,28,28,512) t2, snake scatter to out2.
// One block per destination pixel, 128 threads x float4 = 512 channels.
// ---------------------------------------------------------------------------
__global__ __launch_bounds__(128) void resize_snake_kernel(
    const float* __restrict__ t2, float* __restrict__ out2)
{
    int pix = blockIdx.x;                  // 0 .. 64*42*42-1 (destination pixel)
    int b = pix / (42 * 42);
    int q = pix - b * 42 * 42;
    int oy = q / 42, ox = q - oy * 42;
    int cc = (oy & 1) ? (41 - ox) : ox;    // source column in ex for this dest

    float ys = (float)oy * (27.0f / 41.0f);
    float xs = (float)cc * (27.0f / 41.0f);
    int y0 = (int)floorf(ys); int y1 = (y0 + 1 < 27) ? y0 + 1 : 27;
    int x0 = (int)floorf(xs); int x1 = (x0 + 1 < 27) ? x0 + 1 : 27;
    float wy = ys - (float)y0, wx = xs - (float)x0;

    const float4* r00 = (const float4*)(t2 + (((long)b * 28 + y0) * 28 + x0) * DCH);
    const float4* r01 = (const float4*)(t2 + (((long)b * 28 + y0) * 28 + x1) * DCH);
    const float4* r10 = (const float4*)(t2 + (((long)b * 28 + y1) * 28 + x0) * DCH);
    const float4* r11 = (const float4*)(t2 + (((long)b * 28 + y1) * 28 + x1) * DCH);

    int d = threadIdx.x;
    float4 a00 = r00[d], a01 = r01[d], a10 = r10[d], a11 = r11[d];
    float w00 = (1.0f - wy) * (1.0f - wx), w01 = (1.0f - wy) * wx;
    float w10 = wy * (1.0f - wx),          w11 = wy * wx;
    float4 o;
    o.x = a00.x * w00 + a01.x * w01 + a10.x * w10 + a11.x * w11;
    o.y = a00.y * w00 + a01.y * w01 + a10.y * w10 + a11.y * w11;
    o.z = a00.z * w00 + a01.z * w01 + a10.z * w10 + a11.z * w11;
    o.w = a00.w * w00 + a01.w * w01 + a10.w * w10 + a11.w * w11;
    ((float4*)(out2 + ((long)b * 1764 + oy * 42 + ox) * DCH))[d] = o;
}

// ---------------------------------------------------------------------------
extern "C" void kernel_launch(void* const* d_in, const int* in_sizes, int n_in,
                              void* d_out, int out_size, void* d_ws, size_t ws_size,
                              hipStream_t stream) {
    const float* images   = (const float*)d_in[0];
    const float* w_deform = (const float*)d_in[1];
    const float* w_p0     = (const float*)d_in[2];
    const float* b_p0     = (const float*)d_in[3];
    const float* w_p1     = (const float*)d_in[4];
    const float* b_p1     = (const float*)d_in[5];
    const float* w_ov     = (const float*)d_in[6];
    const float* b_ov     = (const float*)d_in[7];
    float* out = (float*)d_out;

    // workspace carve-up (256B aligned)
    size_t off = 0;
    auto carve = [&](size_t bytes) { size_t o = off; off = (off + bytes + 255) & ~(size_t)255; return o; };
    char* ws = (char*)d_ws;
    _Float16* g   = (_Float16*)(ws + carve((size_t)64 * 3 * IMGW * IMGW * 2));
    _Float16* gp  = (_Float16*)(ws + carve((size_t)64 * 3 * PADW * PADW * 2));
    _Float16* wf0 = (_Float16*)(ws + carve((size_t)DCH * KTOT * 2));
    _Float16* wf1 = (_Float16*)(ws + carve((size_t)DCH * KTOT * 2));
    _Float16* wf2 = (_Float16*)(ws + carve((size_t)DCH * KTOT * 2));
    float*    t2  = (float*)   (ws + carve((size_t)64 * 784 * DCH * 4));

    // 1) guided conv + f16 conversion (plain + padded layouts)
    long padElems = 64L * 3 * PADW * PADW;                 // 10,334,208
    deform_conv_kernel<<<(int)((padElems + 255) / 256), 256, 0, stream>>>(images, w_deform, g, gp);

    // 2) weight conversion
    wcvt_kernel<<<(DCH * KTOT + 255) / 256, 256, 0, stream>>>(w_p0, w_p1, w_ov, wf0, wf1, wf2);

    // 3) GEMMs: 32x32 tiles, 4 waves/block sharing the B tile (all tiles full)
    float* out0 = out;
    float* out1 = out + (size_t)64 * 196 * DCH;            //  6,422,528
    float* out2 = out + (size_t)2 * 64 * 196 * DCH;        // 12,845,056

    // mode 0/1: M=12544 -> 392 m-tiles of 32 -> 98 groups of 4 waves; 16 n-tiles
    int blocks01 = (64 * 196 / 32 / 4) * (DCH / 32);       // 98*16 = 1568
    patch_gemm_kernel<<<blocks01, 128, 0, stream>>>(g, wf0, b_p0, out0, IMGW, 16, 14, 14, 0);
    patch_gemm_kernel<<<blocks01, 128, 0, stream>>>(g, wf1, b_p1, out1, IMGW, 16, 14, 14, 1);

    // mode 2: M=50176 -> 1568 m-tiles of 32 -> 392 groups; 16 n-tiles
    int blocks2 = (64 * 784 / 32 / 4) * (DCH / 32);        // 392*16 = 6272
    patch_gemm_kernel<<<blocks2, 128, 0, stream>>>(gp, wf2, b_ov, t2, PADW, 8, 28, 28, 2);

    // 4) bilinear resize + snake scatter
    resize_snake_kernel<<<64 * 42 * 42, 128, 0, stream>>>(t2, out2);
}
